// SharedSelfAttentionCLSModel_7447473292076
// MI455X (gfx1250) — compile-verified, matrix operand order checked by reading, not run on hardware
//
#include <hip/hip_runtime.h>

// ---------------- model constants ----------------
constexpr int kB   = 32;     // batch
constexpr int kTOK = 64;     // token encoder dim
constexpr int kD   = 256;    // d_model
constexpr int kH   = 8;      // heads
constexpr int kHD  = 32;     // head dim
constexpr int kNL  = 4;      // layers
constexpr int kNT  = 456;    // tokens
constexpr int kS   = 457;    // sequence incl CLS
constexpr int kSP  = 464;    // padded sequence (mult of 16)
constexpr int kM   = kB * kSP;        // 14848 rows (x-space)
constexpr int kMT  = kM / 16;         // 928 row tiles
constexpr int kPOOL = 32;

typedef __attribute__((ext_vector_type(16))) _Float16 v16h;
typedef __attribute__((ext_vector_type(8)))  float    v8f;

union FragU { uint4 q[2]; v16h v; };

__device__ __forceinline__ v8f wmma16(v16h a, v16h b, v8f c) {
  return __builtin_amdgcn_wmma_f32_16x16x32_f16(
      false, a, false, b, (short)0, c, false, false);
}

// Load a 16x32 f16 WMMA fragment (A layout; B uses same layout with "row"=column).
// base points at element (0, 0), the tile starts at (row0, k0); stride = row stride in elements.
// Per ISA 16-bit A layout: lane L (r=L&15, h=L>>4): elems 0..7 <- K=h*8..h*8+7,
// elems 8..15 <- K=16+h*8.. (both 16B-contiguous).
__device__ __forceinline__ v16h load_frag(const _Float16* __restrict__ base,
                                          int stride, int row0, int k0, int lane) {
  int r = lane & 15, h = lane >> 4;
  const _Float16* p = base + (size_t)(row0 + r) * stride + k0 + h * 8;
  FragU u;
  u.q[0] = *(const uint4*)(p);
  u.q[1] = *(const uint4*)(p + 16);
  return u.v;
}

__device__ __forceinline__ float redmax16(float v) {
  v = fmaxf(v, __shfl_xor(v, 1, 32));
  v = fmaxf(v, __shfl_xor(v, 2, 32));
  v = fmaxf(v, __shfl_xor(v, 4, 32));
  v = fmaxf(v, __shfl_xor(v, 8, 32));
  return v;
}
__device__ __forceinline__ float redsum16(float v) {
  v += __shfl_xor(v, 1, 32);
  v += __shfl_xor(v, 2, 32);
  v += __shfl_xor(v, 4, 32);
  v += __shfl_xor(v, 8, 32);
  return v;
}
__device__ __forceinline__ float redsum32(float v) {
  v = redsum16(v);
  v += __shfl_xor(v, 16, 32);
  return v;
}

// ---------------- conversion kernels ----------------
__global__ void conv_f16(const float* __restrict__ in, _Float16* __restrict__ out, int n) {
  int i = blockIdx.x * 256 + threadIdx.x;
  if (i < n) out[i] = (_Float16)in[i];
}

// in: L stacked (R x C) f32 row-major; out: L stacked (C x R) f16 row-major (W^T).
__global__ void tconv_f16(const float* __restrict__ in, _Float16* __restrict__ out,
                          int L, int R, int C) {
  int i = blockIdx.x * 256 + threadIdx.x;
  int tot = L * R * C;
  if (i >= tot) return;
  int l = i / (R * C);
  int rem = i - l * R * C;
  int r = rem / C;
  int c = rem - r * C;
  out[(size_t)l * R * C + (size_t)c * R + r] = (_Float16)in[i];
}

// CLS row (s=0) + zero pad rows (457..463) of x.
__global__ void cls_pad_kernel(const float* __restrict__ coords,
                               const float* __restrict__ Wc,
                               const float* __restrict__ bc,
                               _Float16* __restrict__ x) {
  int b = blockIdx.x, d = threadIdx.x;
  float v = bc[d];
#pragma unroll
  for (int k = 0; k < 3; k++) v += coords[b * 3 + k] * Wc[k * kD + d];
  x[(size_t)b * kSP * kD + d] = (_Float16)v;
  for (int s = kS; s < kSP; s++)
    x[((size_t)b * kSP + s) * kD + d] = (_Float16)0.0f;
}

// ---------------- WMMA GEMM, 32x64 tile per wave: C = A(MxK) * Wt(NxK)^T + bias ----------
// grid.x = M/32, grid.y = N/64. Register-blocked 2x4 WMMA tiles: 8 WMMAs per
// k-step fed by 6 fragment loads (A reused across 4 N-tiles, B across 2 M-tiles).
__global__ __launch_bounds__(32)
void gemm_wmma(const _Float16* __restrict__ A, int lda,
               const _Float16* __restrict__ Wt,
               const float* __restrict__ bias,
               _Float16* __restrict__ C, int ldc,
               int K, int relu) {
  int tm = blockIdx.x * 32;
  int tn = blockIdx.y * 64;
  int lane = threadIdx.x;
  v8f acc[2][4] = {};
  for (int k0 = 0; k0 < K; k0 += 32) {
    v16h a0 = load_frag(A, lda, tm,      k0, lane);
    v16h a1 = load_frag(A, lda, tm + 16, k0, lane);
    v16h b0 = load_frag(Wt, K, tn,      k0, lane);
    v16h b1 = load_frag(Wt, K, tn + 16, k0, lane);
    v16h b2 = load_frag(Wt, K, tn + 32, k0, lane);
    v16h b3 = load_frag(Wt, K, tn + 48, k0, lane);
    acc[0][0] = wmma16(a0, b0, acc[0][0]);
    acc[0][1] = wmma16(a0, b1, acc[0][1]);
    acc[0][2] = wmma16(a0, b2, acc[0][2]);
    acc[0][3] = wmma16(a0, b3, acc[0][3]);
    acc[1][0] = wmma16(a1, b0, acc[1][0]);
    acc[1][1] = wmma16(a1, b1, acc[1][1]);
    acc[1][2] = wmma16(a1, b2, acc[1][2]);
    acc[1][3] = wmma16(a1, b3, acc[1][3]);
  }
  int l16 = lane & 15;
  int half = lane >> 4;
#pragma unroll
  for (int ni = 0; ni < 4; ni++) {
    int n = tn + ni * 16 + l16;
    float bb = bias[n];
#pragma unroll
    for (int mi = 0; mi < 2; mi++) {
#pragma unroll
      for (int i = 0; i < 8; i++) {
        float v = acc[mi][ni][i] + bb;
        if (relu) v = fmaxf(v, 0.0f);
        C[(size_t)(tm + mi * 16 + i + 8 * half) * ldc + n] = (_Float16)v;
      }
    }
  }
}

// Input projection GEMM: A = gene_f16 (B*NT x 64); store remapped into x (b, s=t+1).
__global__ __launch_bounds__(32)
void gemm_inproj(const _Float16* __restrict__ A,
                 const _Float16* __restrict__ Wt,   // (256 x 64)
                 const float* __restrict__ bias,
                 _Float16* __restrict__ X) {
  int tm = blockIdx.x, tn = blockIdx.y;
  int lane = threadIdx.x;
  int row0 = tm * 16, col0 = tn * 16;
  v8f acc = {};
  for (int k0 = 0; k0 < kTOK; k0 += 32) {
    v16h a = load_frag(A, kTOK, row0, k0, lane);
    v16h b = load_frag(Wt, kTOK, col0, k0, lane);
    acc = wmma16(a, b, acc);
  }
  int n = col0 + (lane & 15);
  int half = lane >> 4;
  float bb = bias[n];
#pragma unroll
  for (int i = 0; i < 8; i++) {
    int g = row0 + i + 8 * half;           // global token index in [0, B*NT)
    int b = g / kNT;
    int s = g - b * kNT + 1;               // +1 for CLS
    X[((size_t)b * kSP + s) * kD + n] = (_Float16)(acc[i] + bb);
  }
}

// ---------------- fused flash attention (one wave per (qtile, head, batch)) ----------------
__global__ __launch_bounds__(32)
void attn_kernel(const _Float16* __restrict__ qkv,   // (B*SP, 768)
                 _Float16* __restrict__ out,          // (B*SP, 256)
                 const float* __restrict__ slopes) {
  int qt = blockIdx.x;      // 0..28
  int h  = blockIdx.y;      // 0..7
  int b  = blockIdx.z;      // 0..31
  int lane = threadIdx.x;
  int l16 = lane & 15, hh = lane >> 4;

  __shared__ __align__(16) _Float16 Pt[16 * 32];
  __shared__ __align__(16) _Float16 Vt[32 * 32];

  const _Float16* base = qkv + (size_t)b * kSP * 768;
  int q0 = qt * 16;
  float slope = slopes[h];
  const float scale = 0.1767766952966369f;   // 1/sqrt(32)

  v16h qa = load_frag(base + h * kHD, 768, q0, 0, lane);   // Q fragment (16x32)

  float m_run[8], l_run[8];
  v8f o0 = {}, o1 = {};
#pragma unroll
  for (int i = 0; i < 8; i++) { m_run[i] = -1e30f; l_run[i] = 0.0f; }

  for (int jt = 0; jt < 15; jt++) {          // 15*32 = 480 keys (masked past 456)
    int kbase = jt * 32;
    // K^T fragments: lane column = key index, K-dim = head dim (contiguous in row)
    int key0 = kbase + l16;       if (key0 > kSP - 1) key0 = kSP - 1;
    int key1 = kbase + 16 + l16;  if (key1 > kSP - 1) key1 = kSP - 1;
    FragU ku0, ku1;
    {
      const _Float16* p = base + (size_t)key0 * 768 + kD + h * kHD + hh * 8;
      ku0.q[0] = *(const uint4*)p; ku0.q[1] = *(const uint4*)(p + 16);
      p = base + (size_t)key1 * 768 + kD + h * kHD + hh * 8;
      ku1.q[0] = *(const uint4*)p; ku1.q[1] = *(const uint4*)(p + 16);
    }
    v8f zero = {};
    v8f s0 = wmma16(qa, ku0.v, zero);
    v8f s1 = wmma16(qa, ku1.v, zero);

    __syncthreads();
    // V tile (32 keys x 32 dims) -> LDS, one row per lane
    {
      int key = kbase + lane; if (key > kSP - 1) key = kSP - 1;
      const uint4* vp = (const uint4*)(base + (size_t)key * 768 + 2 * kD + h * kHD);
      uint4* vs = (uint4*)(Vt + lane * 32);
      vs[0] = vp[0]; vs[1] = vp[1]; vs[2] = vp[2]; vs[3] = vp[3];
    }

    // online softmax per row (row = i + 8*hh lives across the 16 lanes of this half)
#pragma unroll
    for (int i = 0; i < 8; i++) {
      int sq  = q0 + i + 8 * hh;
      int sk0 = kbase + l16;
      int sk1 = sk0 + 16;
      float v0 = s0[i] * scale - slope * fabsf((float)(sq - sk0));
      float v1 = s1[i] * scale - slope * fabsf((float)(sq - sk1));
      if (sk0 >= kS) v0 = -1e30f;
      if (sk1 >= kS) v1 = -1e30f;
      float tmax = redmax16(fmaxf(v0, v1));
      float nm   = fmaxf(m_run[i], tmax);
      float coef = __expf(m_run[i] - nm);
      float p0 = __expf(v0 - nm), p1 = __expf(v1 - nm);
      float ps = redsum16(p0 + p1);
      l_run[i] = l_run[i] * coef + ps;
      m_run[i] = nm;
      o0[i] *= coef;
      o1[i] *= coef;
      Pt[(i + 8 * hh) * 32 + l16]      = (_Float16)p0;
      Pt[(i + 8 * hh) * 32 + l16 + 16] = (_Float16)p1;
    }
    __syncthreads();

    // Re-fragment P (16x32) from LDS into A layout
    FragU pu;
    pu.q[0] = *(const uint4*)(Pt + l16 * 32 + hh * 8);
    pu.q[1] = *(const uint4*)(Pt + l16 * 32 + 16 + hh * 8);
    // V B-fragments from LDS (column = head dim)
    v16h vb0, vb1;
#pragma unroll
    for (int e = 0; e < 16; e++) {
      int k = (e < 8) ? (hh * 8 + e) : (16 + hh * 8 + (e - 8));
      vb0[e] = Vt[k * 32 + l16];
      vb1[e] = Vt[k * 32 + l16 + 16];
    }
    o0 = wmma16(pu.v, vb0, o0);
    o1 = wmma16(pu.v, vb1, o1);
  }

#pragma unroll
  for (int i = 0; i < 8; i++) {
    float inv = 1.0f / l_run[i];
    size_t row = (size_t)b * kSP + q0 + i + 8 * hh;
    out[row * kD + h * kHD + l16]      = (_Float16)(o0[i] * inv);
    out[row * kD + h * kHD + l16 + 16] = (_Float16)(o1[i] * inv);
  }
}

// ---------------- residual + LayerNorm (in place on x): x = LN(x + y) ----------------
__global__ void ln_kernel(_Float16* __restrict__ x, const _Float16* __restrict__ y,
                          const float* __restrict__ g, const float* __restrict__ bt) {
  int row  = blockIdx.x * 8 + (threadIdx.x >> 5);
  int lane = threadIdx.x & 31;
  _Float16* xr = x + (size_t)row * kD;
  const _Float16* yr = y + (size_t)row * kD;
  float v[8];
  float s = 0.0f;
#pragma unroll
  for (int j = 0; j < 8; j++) {
    int c = lane + j * 32;
    float t = (float)xr[c] + (float)yr[c];
    v[j] = t; s += t;
  }
  s = redsum32(s);
  float mean = s * (1.0f / kD);
  float var = 0.0f;
#pragma unroll
  for (int j = 0; j < 8; j++) { float d = v[j] - mean; var += d * d; }
  var = redsum32(var) * (1.0f / kD);
  float rstd = rsqrtf(var + 1e-5f);
#pragma unroll
  for (int j = 0; j < 8; j++) {
    int c = lane + j * 32;
    xr[c] = (_Float16)((v[j] - mean) * rstd * g[c] + bt[c]);
  }
}

// ---------------- attention pooling (one block per batch element) ----------------
__global__ void pool_kernel(const _Float16* __restrict__ x,
                            const float* __restrict__ p1w, const float* __restrict__ p1b,
                            const float* __restrict__ bng, const float* __restrict__ bnb,
                            const float* __restrict__ p2w, const float* __restrict__ p2b,
                            float* __restrict__ pooled) {
  int b = blockIdx.x, tid = threadIdx.x;
  __shared__ float sc[kS];
  __shared__ float red[256];
  const _Float16* xb = x + (size_t)b * kSP * kD;
  float bnscale = rsqrtf(1.0f + 1e-5f);

  for (int s = tid; s < kS; s += 256) {
    const _Float16* xr = xb + (size_t)s * kD;
    float score = p2b[0];
    for (int hh = 0; hh < kPOOL; hh++) {
      float acc = p1b[hh];
      for (int d = 0; d < kD; d++) acc += (float)xr[d] * p1w[d * kPOOL + hh];
      acc = acc * bnscale * bng[hh] + bnb[hh];
      float ge = 0.5f * acc * (1.0f + erff(acc * 0.7071067811865475f));
      score += ge * p2w[hh];
    }
    sc[s] = score;
  }
  __syncthreads();
  float mx = -1e30f;
  for (int s = tid; s < kS; s += 256) mx = fmaxf(mx, sc[s]);
  red[tid] = mx; __syncthreads();
  for (int o = 128; o > 0; o >>= 1) { if (tid < o) red[tid] = fmaxf(red[tid], red[tid + o]); __syncthreads(); }
  mx = red[0]; __syncthreads();
  float sm = 0.0f;
  for (int s = tid; s < kS; s += 256) { float e = __expf(sc[s] - mx); sc[s] = e; sm += e; }
  red[tid] = sm; __syncthreads();
  for (int o = 128; o > 0; o >>= 1) { if (tid < o) red[tid] += red[tid + o]; __syncthreads(); }
  float inv = 1.0f / red[0];
  int d = tid;
  float acc = 0.0f;
  for (int s = 0; s < kS; s++) acc += sc[s] * (float)xb[(size_t)s * kD + d];
  pooled[(size_t)b * kD + d] = acc * inv;
}

// ---------------- final head: (B, 2D) @ (2D,1) + b ----------------
__global__ void head_kernel(const float* __restrict__ pooled,
                            const float* __restrict__ Wh, const float* __restrict__ bh,
                            float* __restrict__ out) {
  int b = threadIdx.x;
  float acc = bh[0];
  for (int k = 0; k < kD; k++) acc += pooled[b * kD + k] * Wh[k];
  for (int k = 0; k < kD; k++) acc += pooled[(kB + b) * kD + k] * Wh[kD + k];
  out[b] = acc;
}

// ---------------- host launch ----------------
extern "C" void kernel_launch(void* const* d_in, const int* in_sizes, int n_in,
                              void* d_out, int out_size, void* d_ws, size_t ws_size,
                              hipStream_t stream) {
  const float* gene_i   = (const float*)d_in[0];
  const float* coords_i = (const float*)d_in[1];
  const float* gene_j   = (const float*)d_in[2];
  const float* coords_j = (const float*)d_in[3];
  const float* W_in     = (const float*)d_in[4];
  const float* b_in     = (const float*)d_in[5];
  const float* W_cls    = (const float*)d_in[6];
  const float* b_cls    = (const float*)d_in[7];
  const float* qkv_w    = (const float*)d_in[8];
  const float* qkv_b    = (const float*)d_in[9];
  const float* ln1_g    = (const float*)d_in[10];
  const float* ln1_b    = (const float*)d_in[11];
  const float* ffn_w1   = (const float*)d_in[12];
  const float* ffn_b1   = (const float*)d_in[13];
  const float* ffn_w2   = (const float*)d_in[14];
  const float* ffn_b2   = (const float*)d_in[15];
  const float* ln2_g    = (const float*)d_in[16];
  const float* ln2_b    = (const float*)d_in[17];
  const float* p1_w     = (const float*)d_in[18];
  const float* p1_b     = (const float*)d_in[19];
  const float* bn_g     = (const float*)d_in[20];
  const float* bn_b     = (const float*)d_in[21];
  const float* p2_w     = (const float*)d_in[22];
  const float* p2_b     = (const float*)d_in[23];
  const float* W_head   = (const float*)d_in[24];
  const float* b_head   = (const float*)d_in[25];
  const float* slopes   = (const float*)d_in[26];

  char* ws = (char*)d_ws;
  size_t off = 0;
  auto take = [&](size_t bytes) -> void* {
    void* p = ws + off;
    off = (off + bytes + 255) & ~(size_t)255;
    return p;
  };
  _Float16* gf16   = (_Float16*)take((size_t)kB * kNT * kTOK * 2);
  _Float16* Wt_in  = (_Float16*)take((size_t)kD * kTOK * 2);
  _Float16* Wt_qkv = (_Float16*)take((size_t)kNL * 3 * kD * kD * 2);
  _Float16* Wt_f1  = (_Float16*)take((size_t)kNL * 4 * kD * kD * 2);
  _Float16* Wt_f2  = (_Float16*)take((size_t)kNL * 4 * kD * kD * 2);
  _Float16* xbuf   = (_Float16*)take((size_t)kM * kD * 2);
  _Float16* qkvbuf = (_Float16*)take((size_t)kM * 3 * kD * 2);
  _Float16* ybuf   = (_Float16*)take((size_t)kM * kD * 2);
  _Float16* hbuf   = (_Float16*)take((size_t)kM * 4 * kD * 2);
  float*    pooled = (float*)take((size_t)2 * kB * kD * 4);

  // one-time weight conversion/transposition (f32 -> f16, (R,C) -> (C,R))
  {
    int n = 1 * kTOK * kD;
    tconv_f16<<<(n + 255) / 256, 256, 0, stream>>>(W_in, Wt_in, 1, kTOK, kD);
    n = kNL * kD * 3 * kD;
    tconv_f16<<<(n + 255) / 256, 256, 0, stream>>>(qkv_w, Wt_qkv, kNL, kD, 3 * kD);
    n = kNL * kD * 4 * kD;
    tconv_f16<<<(n + 255) / 256, 256, 0, stream>>>(ffn_w1, Wt_f1, kNL, kD, 4 * kD);
    n = kNL * 4 * kD * kD;
    tconv_f16<<<(n + 255) / 256, 256, 0, stream>>>(ffn_w2, Wt_f2, kNL, 4 * kD, kD);
  }

  for (int enc = 0; enc < 2; enc++) {
    const float* gene   = enc ? gene_j : gene_i;
    const float* coords = enc ? coords_j : coords_i;

    int n = kB * kNT * kTOK;
    conv_f16<<<(n + 255) / 256, 256, 0, stream>>>(gene, gf16, n);
    cls_pad_kernel<<<kB, kD, 0, stream>>>(coords, W_cls, b_cls, xbuf);
    gemm_inproj<<<dim3(kB * kNT / 16, kD / 16), 32, 0, stream>>>(gf16, Wt_in, b_in, xbuf);

    for (int l = 0; l < kNL; l++) {
      gemm_wmma<<<dim3(kMT / 2, 3 * kD / 64), 32, 0, stream>>>(
          xbuf, kD, Wt_qkv + (size_t)l * 3 * kD * kD, qkv_b + l * 3 * kD,
          qkvbuf, 3 * kD, kD, 0);
      attn_kernel<<<dim3(29, kH, kB), 32, 0, stream>>>(qkvbuf, ybuf, slopes);
      ln_kernel<<<kM / 8, 256, 0, stream>>>(xbuf, ybuf, ln1_g + l * kD, ln1_b + l * kD);
      gemm_wmma<<<dim3(kMT / 2, 4 * kD / 64), 32, 0, stream>>>(
          xbuf, kD, Wt_f1 + (size_t)l * 4 * kD * kD, ffn_b1 + l * 4 * kD,
          hbuf, 4 * kD, kD, 1);
      gemm_wmma<<<dim3(kMT / 2, kD / 64), 32, 0, stream>>>(
          hbuf, 4 * kD, Wt_f2 + (size_t)l * 4 * kD * kD, ffn_b2 + l * kD,
          ybuf, kD, 4 * kD, 0);
      ln_kernel<<<kM / 8, 256, 0, stream>>>(xbuf, ybuf, ln2_g + l * kD, ln2_b + l * kD);
    }
    pool_kernel<<<kB, 256, 0, stream>>>(xbuf, p1_w, p1_b, bn_g, bn_b, p2_w, p2_b,
                                        pooled + (size_t)enc * kB * kD);
  }
  head_kernel<<<1, kB, 0, stream>>>(pooled, W_head, b_head, (float*)d_out);
}
